// ProductManifoldPhasorBlock_37082747633871
// MI455X (gfx1250) — compile-verified, matrix operand order checked by reading, not run on hardware
//
#include <hip/hip_runtime.h>
#include <hip/hip_bf16.h>
#include <math.h>

#define B_   4
#define L_   1024
#define D_   256
#define NPER 48
#define NHYP 16
#define KOSC 64   // NPER + NHYP

typedef __attribute__((ext_vector_type(16))) __bf16 v16bf;
typedef __attribute__((ext_vector_type(8)))  float  v8f;

union FragAB { v16bf v; unsigned short u[16]; };
union FragC  { v8f   v; float          f[8];  };

__device__ __forceinline__ unsigned short f32_to_bf16(float x) {
    unsigned int u = __float_as_uint(x);
    u += 0x7FFFu + ((u >> 16) & 1u);   // round-to-nearest-even
    return (unsigned short)(u >> 16);
}

// ---------------------------------------------------------------- conversions
__global__ void cvt_f32_bf16_kernel(const float* __restrict__ in,
                                    unsigned short* __restrict__ out, int n) {
    int i = blockIdx.x * blockDim.x + threadIdx.x;
    if (i < n) out[i] = f32_to_bf16(in[i]);
}

// V: (B,L,D) f32 -> Vt: (B,D,L) bf16
__global__ void transpose_cvt_kernel(const float* __restrict__ V,
                                     unsigned short* __restrict__ Vt, int n) {
    int i = blockIdx.x * blockDim.x + threadIdx.x;
    if (i >= n) return;
    int d = i % D_;
    int l = (i / D_) % L_;
    int b = i / (D_ * L_);
    Vt[((long)b * D_ + d) * L_ + l] = f32_to_bf16(V[i]);
}

// ------------------------------------------------------- batched WMMA GEMM
// C[b,m,n] = sum_k A[b,m,k] * Bm[b,n,k]  (+ bias[n]) (+ resid) ; act==1 -> GELU
// A, Bm bf16 (row-major, K contiguous). Each wave computes a 16(M) x 64(N)
// strip: one A fragment feeds 4 independent WMMA accumulators per k-step.
// N tiles are clamped (never masked) so EXEC stays all-ones around WMMA.
__global__ __launch_bounds__(128)
void gemm_bf16_wmma(const unsigned short* __restrict__ A, int lda, long sA,
                    const unsigned short* __restrict__ Bm, int ldb, long sB,
                    float* __restrict__ C, int ldc, long sC,
                    const float* __restrict__ bias,
                    const float* __restrict__ resid, long sR,
                    int M, int N, int K, int act)
{
    const int wave = threadIdx.x >> 5;
    const int lane = threadIdx.x & 31;
    const int m0   = blockIdx.x * 64 + wave * 16;
    if (m0 >= M) return;                      // wave-uniform
    const int b = blockIdx.z;

    const unsigned short* Ab = A  + (long)b * sA;
    const unsigned short* Bb = Bm + (long)b * sB;
    float*                Cb = C  + (long)b * sC;

    const int lm = lane & 15;
    const int kb = (lane < 16) ? 0 : 8;       // K sub-block per lane half

    const unsigned short* arow = Ab + (long)(m0 + lm) * lda + kb;

    const int tmax = N / 16 - 1;              // N is a multiple of 16
    int nt[4];
    const unsigned short* brow[4];
    #pragma unroll
    for (int j = 0; j < 4; ++j) {
        int t = blockIdx.y * 4 + j;
        nt[j] = (t < tmax) ? t : tmax;        // clamp: duplicates are benign
        brow[j] = Bb + (long)(nt[j] * 16 + lm) * ldb + kb;
    }

    FragC acc[4];
    #pragma unroll
    for (int j = 0; j < 4; ++j)
        #pragma unroll
        for (int i = 0; i < 8; ++i) acc[j].f[i] = 0.0f;

    for (int k0 = 0; k0 < K; k0 += 32) {
        FragAB a;
        #pragma unroll
        for (int e = 0; e < 8; ++e) {
            a.u[e]     = arow[k0 + e];
            a.u[e + 8] = arow[k0 + 16 + e];
        }
        __builtin_prefetch(arow + k0 + 64, 0, 1);
        #pragma unroll
        for (int j = 0; j < 4; ++j) {
            FragAB bf;
            #pragma unroll
            for (int e = 0; e < 8; ++e) {
                bf.u[e]     = brow[j][k0 + e];
                bf.u[e + 8] = brow[j][k0 + 16 + e];
            }
            __builtin_prefetch(brow[j] + k0 + 64, 0, 1);
            acc[j].v = __builtin_amdgcn_wmma_f32_16x16x32_bf16(
                           false, a.v, false, bf.v, (short)0, acc[j].v,
                           false, false);
        }
    }

    const int rbase = m0 + ((lane >> 4) << 3);  // C: row = m0 + v + 8*(lane/16)
    const float* Rb = resid ? (resid + (long)b * sR) : nullptr;
    #pragma unroll
    for (int j = 0; j < 4; ++j) {
        const int   col = nt[j] * 16 + lm;
        const float bv  = bias ? bias[col] : 0.0f;
        #pragma unroll
        for (int v = 0; v < 8; ++v) {
            const long idx = (long)(rbase + v) * ldc + col;
            float val = acc[j].f[v] + bv;
            if (act == 1) val = 0.5f * val * (1.0f + erff(val * 0.70710678118654752f));
            if (Rb) val += Rb[idx];
            Cb[idx] = val;
        }
    }
}

// ------------------------------------------------------------------ phasors
// Build Qcat/Kcat (B*L, 128) bf16 = [Re(64) | Im(64)] so that
// Qcat . Kcat^T = Qr.Kr + Qi.Ki = Re(conj(q) * k).
__global__ void phasor_kernel(const float* __restrict__ qp, const float* __restrict__ kp,
                              const float* __restrict__ qh, const float* __restrict__ kh,
                              unsigned short* __restrict__ Qcat,
                              unsigned short* __restrict__ Kcat, int total)
{
    int idx = blockIdx.x * blockDim.x + threadIdx.x;
    int t = idx / KOSC;
    int k = idx % KOSC;
    if (t >= total) return;

    const float PI = 3.14159265358979323846f;
    float qr, qi, kr, ki;
    if (k < NPER) {
        float th = tanhf(qp[t * NPER + k]) * PI;
        qr = cosf(th); qi = sinf(th);
        th = tanhf(kp[t * NPER + k]) * PI;
        kr = cosf(th); ki = sinf(th);
    } else {
        int j = k - NPER;
        float a = tanhf(qh[t * (2 * NHYP) + 2 * j]) * 0.9f;
        float c = tanhf(qh[t * (2 * NHYP) + 2 * j + 1]) * 0.9f;
        float r = sqrtf(a * a + c * c);
        float amp = 1.0f - r;
        if (r > 1e-12f) { qr = amp * a / r; qi = amp * c / r; }
        else            { qr = amp;         qi = 0.0f;        }
        a = tanhf(kh[t * (2 * NHYP) + 2 * j]) * 0.9f;
        c = tanhf(kh[t * (2 * NHYP) + 2 * j + 1]) * 0.9f;
        r = sqrtf(a * a + c * c);
        amp = 1.0f - r;
        if (r > 1e-12f) { kr = amp * a / r; ki = amp * c / r; }
        else            { kr = amp;         ki = 0.0f;        }
    }
    Qcat[(long)t * 128 + k]      = f32_to_bf16(qr);
    Qcat[(long)t * 128 + 64 + k] = f32_to_bf16(qi);
    Kcat[(long)t * 128 + k]      = f32_to_bf16(kr);
    Kcat[(long)t * 128 + 64 + k] = f32_to_bf16(ki);
}

// causal mask + fold in 1/sqrt((t+1)*K) row scaling; emits bf16 directly
__global__ void mask_scale_bf16_kernel(const float* __restrict__ S,
                                       unsigned short* __restrict__ Sbf) {
    long idx = (long)blockIdx.x * blockDim.x + threadIdx.x;
    long tot = (long)B_ * L_ * L_;
    if (idx >= tot) return;
    int s = (int)(idx % L_);
    int t = (int)((idx / L_) % L_);
    float v = S[idx];
    if (s > t) v = 0.0f;
    else       v *= rsqrtf((float)(t + 1) * (float)KOSC);
    Sbf[idx] = f32_to_bf16(v);
}

// -------------------------------------------------------------- LayerNorm
__global__ __launch_bounds__(256)
void ln_kernel(const float* __restrict__ R, const float* __restrict__ g,
               const float* __restrict__ bb, unsigned short* __restrict__ out)
{
    __shared__ float red[256];
    int row = blockIdx.x;
    int d   = threadIdx.x;
    float v = R[(long)row * D_ + d];
    red[d] = v; __syncthreads();
    for (int off = 128; off > 0; off >>= 1) {
        if (d < off) red[d] += red[d + off];
        __syncthreads();
    }
    float mu = red[0] * (1.0f / D_); __syncthreads();
    float c = v - mu;
    red[d] = c * c; __syncthreads();
    for (int off = 128; off > 0; off >>= 1) {
        if (d < off) red[d] += red[d + off];
        __syncthreads();
    }
    float var = red[0] * (1.0f / D_);
    float y = c * rsqrtf(var + 1e-5f) * g[d] + bb[d];
    out[(long)row * D_ + d] = f32_to_bf16(y);
}

// ------------------------------------------------------------------- launch
extern "C" void kernel_launch(void* const* d_in, const int* in_sizes, int n_in,
                              void* d_out, int out_size, void* d_ws, size_t ws_size,
                              hipStream_t stream) {
    (void)in_sizes; (void)n_in; (void)out_size; (void)ws_size;

    const float* x     = (const float*)d_in[0];
    const float* kp_w1 = (const float*)d_in[1];  const float* kp_b1 = (const float*)d_in[2];
    const float* kp_w2 = (const float*)d_in[3];  const float* kp_b2 = (const float*)d_in[4];
    const float* qp_w1 = (const float*)d_in[5];  const float* qp_b1 = (const float*)d_in[6];
    const float* qp_w2 = (const float*)d_in[7];  const float* qp_b2 = (const float*)d_in[8];
    const float* kh_w1 = (const float*)d_in[9];  const float* kh_b1 = (const float*)d_in[10];
    const float* kh_w2 = (const float*)d_in[11]; const float* kh_b2 = (const float*)d_in[12];
    const float* qh_w1 = (const float*)d_in[13]; const float* qh_b1 = (const float*)d_in[14];
    const float* qh_w2 = (const float*)d_in[15]; const float* qh_b2 = (const float*)d_in[16];
    const float* v_w   = (const float*)d_in[17]; const float* v_b   = (const float*)d_in[18];
    const float* ln_g  = (const float*)d_in[19]; const float* ln_b  = (const float*)d_in[20];
    const float* out_w = (const float*)d_in[21]; const float* out_b = (const float*)d_in[22];
    float* y = (float*)d_out;

    const int  nBL = B_ * L_;            // 4096
    const int  nX  = nBL * D_;           // 1,048,576
    const long nS  = (long)B_ * L_ * L_; // 4,194,304

    // bump allocator on d_ws
    char*  ws  = (char*)d_ws;
    size_t off = 0;
    auto alloc = [&](size_t bytes) -> void* {
        off = (off + 255) & ~(size_t)255;
        void* p = ws + off;
        off += bytes;
        return p;
    };

    unsigned short* xbf     = (unsigned short*)alloc((size_t)nX * 2);
    unsigned short* kp_w1bf = (unsigned short*)alloc(65536 * 2);
    unsigned short* qp_w1bf = (unsigned short*)alloc(65536 * 2);
    unsigned short* kh_w1bf = (unsigned short*)alloc(65536 * 2);
    unsigned short* qh_w1bf = (unsigned short*)alloc(65536 * 2);
    unsigned short* v_wbf   = (unsigned short*)alloc(65536 * 2);
    unsigned short* out_wbf = (unsigned short*)alloc(65536 * 2);
    unsigned short* kp_w2bf = (unsigned short*)alloc(NPER * D_ * 2);
    unsigned short* qp_w2bf = (unsigned short*)alloc(NPER * D_ * 2);
    unsigned short* kh_w2bf = (unsigned short*)alloc(2 * NHYP * D_ * 2);
    unsigned short* qh_w2bf = (unsigned short*)alloc(2 * NHYP * D_ * 2);
    float*          Hf      = (float*)alloc((size_t)nX * 4);
    unsigned short* Hbf     = (unsigned short*)alloc((size_t)nX * 2);
    float*          kp_o    = (float*)alloc((size_t)nBL * NPER * 4);
    float*          qp_o    = (float*)alloc((size_t)nBL * NPER * 4);
    float*          kh_o    = (float*)alloc((size_t)nBL * 2 * NHYP * 4);
    float*          qh_o    = (float*)alloc((size_t)nBL * 2 * NHYP * 4);
    unsigned short* Qcat    = (unsigned short*)alloc((size_t)nBL * 128 * 2);
    unsigned short* Kcat    = (unsigned short*)alloc((size_t)nBL * 128 * 2);
    float*          Vf      = (float*)alloc((size_t)nX * 4);
    unsigned short* Vtbf    = (unsigned short*)alloc((size_t)nX * 2);
    float*          S       = (float*)alloc((size_t)nS * 4);
    unsigned short* Sbf     = (unsigned short*)alloc((size_t)nS * 2);
    float*          R       = (float*)alloc((size_t)nX * 4);
    unsigned short* rlnbf   = (unsigned short*)alloc((size_t)nX * 2);

    auto cvt = [&](const float* in, unsigned short* out, int n) {
        cvt_f32_bf16_kernel<<<(n + 255) / 256, 256, 0, stream>>>(in, out, n);
    };
    auto gemm = [&](const unsigned short* A, int lda, long sA,
                    const unsigned short* Bm, int ldb, long sB,
                    float* C, int ldc, long sC, const float* bias,
                    const float* resid, long sR,
                    int M, int N, int K, int act, int batch) {
        dim3 grid(M / 64, (N / 16 + 3) / 4, batch);
        gemm_bf16_wmma<<<grid, 128, 0, stream>>>(A, lda, sA, Bm, ldb, sB,
                                                 C, ldc, sC, bias, resid, sR,
                                                 M, N, K, act);
    };

    // -- convert activations / weights to bf16
    cvt(x, xbf, nX);
    cvt(kp_w1, kp_w1bf, 65536); cvt(qp_w1, qp_w1bf, 65536);
    cvt(kh_w1, kh_w1bf, 65536); cvt(qh_w1, qh_w1bf, 65536);
    cvt(v_w, v_wbf, 65536);     cvt(out_w, out_wbf, 65536);
    cvt(kp_w2, kp_w2bf, NPER * D_);     cvt(qp_w2, qp_w2bf, NPER * D_);
    cvt(kh_w2, kh_w2bf, 2 * NHYP * D_); cvt(qh_w2, qh_w2bf, 2 * NHYP * D_);

    // -- four MLPs: GELU(x @ W1^T + b1) @ W2^T + b2
    struct { const unsigned short* w1; const float* b1;
             const unsigned short* w2; const float* b2;
             float* out; int n2; } mlps[4] = {
        { kp_w1bf, kp_b1, kp_w2bf, kp_b2, kp_o, NPER },
        { qp_w1bf, qp_b1, qp_w2bf, qp_b2, qp_o, NPER },
        { kh_w1bf, kh_b1, kh_w2bf, kh_b2, kh_o, 2 * NHYP },
        { qh_w1bf, qh_b1, qh_w2bf, qh_b2, qh_o, 2 * NHYP },
    };
    for (int m = 0; m < 4; ++m) {
        gemm(xbf, D_, 0, mlps[m].w1, D_, 0, Hf, D_, 0, mlps[m].b1,
             nullptr, 0, nBL, D_, D_, /*gelu*/1, 1);
        cvt(Hf, Hbf, nX);
        gemm(Hbf, D_, 0, mlps[m].w2, D_, 0, mlps[m].out, mlps[m].n2, 0, mlps[m].b2,
             nullptr, 0, nBL, mlps[m].n2, D_, 0, 1);
    }

    // -- V projection
    gemm(xbf, D_, 0, v_wbf, D_, 0, Vf, D_, 0, v_b, nullptr, 0,
         nBL, D_, D_, 0, 1);

    // -- phasors -> Qcat/Kcat bf16 [Re|Im]
    {
        int n = nBL * KOSC;
        phasor_kernel<<<(n + 255) / 256, 256, 0, stream>>>(qp_o, kp_o, qh_o, kh_o,
                                                           Qcat, Kcat, nBL);
    }

    // -- scores S = Qcat @ Kcat^T  (batched per B)
    gemm(Qcat, 128, (long)L_ * 128, Kcat, 128, (long)L_ * 128,
         S, L_, (long)L_ * L_, nullptr, nullptr, 0, L_, L_, 128, 0, B_);

    // -- causal mask + 1/sqrt((t+1)*K) scaling, fused bf16 emit
    mask_scale_bf16_kernel<<<(int)((nS + 255) / 256), 256, 0, stream>>>(S, Sbf);

    // -- V transpose (B,L,D)->(B,D,L) bf16
    transpose_cvt_kernel<<<(nX + 255) / 256, 256, 0, stream>>>(Vf, Vtbf, nX);

    // -- retrieved R = S @ V  (batched)
    gemm(Sbf, L_, (long)L_ * L_, Vtbf, L_, (long)D_ * L_,
         R, D_, (long)L_ * D_, nullptr, nullptr, 0, L_, D_, L_, 0, B_);

    // -- LayerNorm -> bf16
    ln_kernel<<<nBL, 256, 0, stream>>>(R, ln_g, ln_b, rlnbf);

    // -- output projection + bias + residual, written straight to d_out
    gemm(rlnbf, D_, 0, out_wbf, D_, 0, y, D_, 0, out_b, x, 0,
         nBL, D_, D_, 0, 1);
}
